// MultiHeadAttention_65481071410942
// MI455X (gfx1250) — matrix operand order repaired and checked by reading, not compiled, not run on hardware
//
#include <hip/hip_runtime.h>
#include <hip/hip_bf16.h>

// ---------------------------------------------------------------------------
// CDNA5 / gfx1250 fused multi-head attention, bf16 WMMA path with
// ASYNC global->LDS staging (ASYNCcnt) and double-buffered LDS tiles.
// Pipelined loops are manually unrolled x2 so the LDS buffer parity is a
// compile-time constant (folds into ds_load offsets; no branchy SALU selects).
//   EMB=768, HEADS=8, HEAD_DIM=96, B=4, N=2048 (M = B*N = 8192 rows)
// ---------------------------------------------------------------------------

typedef __bf16 v16bf __attribute__((ext_vector_type(16)));
typedef __bf16 v8bf  __attribute__((ext_vector_type(8)));
typedef float  v8f   __attribute__((ext_vector_type(8)));

#define EMB      768
#define NHEAD    8
#define HDIM     96
#define BATCH    4
#define SEQ      2048
#define MROWS    (BATCH * SEQ)       // 8192
#define QKV_N    (3 * EMB)           // 2304

__device__ __forceinline__ v8f wmma_bf16(v16bf a, v16bf b, v8f c) {
    // D = A(16x32 bf16) x B(32x16 bf16) + C(16x16 f32)
    return __builtin_amdgcn_wmma_f32_16x16x32_bf16(
        /*neg_a=*/false, a, /*neg_b=*/false, b,
        /*c_mod=*/(short)0, c, /*reuse_a=*/false, /*reuse_b=*/false);
}

// Async 16B copy: global memory -> LDS, tracked by ASYNCcnt (no VGPR bounce).
__device__ __forceinline__ void async_ld_b128(void* lds_dst, const void* gsrc) {
    unsigned lds_off = (unsigned)(unsigned long long)lds_dst;  // DS byte offset
    asm volatile("global_load_async_to_lds_b128 %0, %1, off"
                 :: "v"(lds_off), "v"(gsrc) : "memory");
}
__device__ __forceinline__ void wait_async0() {
    asm volatile("s_wait_asynccnt 0x0" ::: "memory");
}

// ---------------------------------------------------------------------------
// Kernel 0: transpose + convert weights f32 [K][Nc] -> bf16 [Nc][K]
// (so WMMA B-fragments become one contiguous 32B load per lane)
// ---------------------------------------------------------------------------
__global__ void repack_w_kernel(const float* __restrict__ W,
                                __bf16* __restrict__ Wt,
                                int K, int Nc) {
    int i = blockIdx.x * 256 + threadIdx.x;
    if (i >= K * Nc) return;
    int n = i / K;
    int k = i - n * K;
    Wt[(size_t)n * K + k] = (__bf16)W[(size_t)k * Nc + n];
}

// ---------------------------------------------------------------------------
// Kernel 1: QKV GEMM.  acc = X[8192,768] @ Wqkv + b, scattered to
//   Qb[B,H,N,D] bf16, Kb[B,H,N,D] bf16, Vt[B,H,D,N] bf16 (V transposed).
// Block: 256 threads = 8 waves; block tile 128(M) x 128(N); K-step 32.
// B tile double-buffered in LDS, filled with async global->LDS loads.
// ---------------------------------------------------------------------------
__global__ __launch_bounds__(256)
void qkv_gemm_kernel(const float* __restrict__ X,
                     const float* __restrict__ bqkv,
                     const __bf16* __restrict__ Wt,   // [2304][768] bf16
                     __bf16* __restrict__ Qb,
                     __bf16* __restrict__ Kb,
                     __bf16* __restrict__ Vt) {
    __shared__ __align__(32) __bf16 ldsB[2][128 * 32];   // [buf][n][k] 2x8 KB

    const int lane  = threadIdx.x & 31;
    const int wave  = threadIdx.x >> 5;
    const int n0    = blockIdx.x * 128;
    const int m0    = (blockIdx.y * 8 + wave) * 16;
    const int arow  = m0 + (lane & 15);
    const int khalf = (lane >> 4) * 8;    // 0 or 8 (unused)
    const int kh16  = (lane >> 4) * 16;   // 0 or 16 (B-frag K pattern)

    // per-thread staging coordinates: 32B (= 2 x b128) per thread per K-step
    const int sn = threadIdx.x >> 1;
    const int sk = (threadIdx.x & 1) * 16;

    auto stage = [&](int k0, int p) {
        const __bf16* src = Wt + (size_t)(n0 + sn) * EMB + k0 + sk;
        __bf16*       dst = &ldsB[p][sn * 32 + sk];
        async_ld_b128(dst, src);
        async_ld_b128(dst + 8, src + 8);
    };

    v8f acc[8] = {};

    auto compute = [&](int k0, const __bf16* Bt) {
        // A fragment: lane = row, K = {kh16..kh16+15}
        v16bf a;
        const float* ap = X + (size_t)arow * EMB + k0;
        float4 l0 = *(const float4*)(ap + kh16);
        float4 l1 = *(const float4*)(ap + kh16 + 4);
        float4 h0 = *(const float4*)(ap + kh16 + 8);
        float4 h1 = *(const float4*)(ap + kh16 + 12);
        a[0]  = (__bf16)l0.x; a[1]  = (__bf16)l0.y; a[2]  = (__bf16)l0.z; a[3]  = (__bf16)l0.w;
        a[4]  = (__bf16)l1.x; a[5]  = (__bf16)l1.y; a[6]  = (__bf16)l1.z; a[7]  = (__bf16)l1.w;
        a[8]  = (__bf16)h0.x; a[9]  = (__bf16)h0.y; a[10] = (__bf16)h0.z; a[11] = (__bf16)h0.w;
        a[12] = (__bf16)h1.x; a[13] = (__bf16)h1.y; a[14] = (__bf16)h1.z; a[15] = (__bf16)h1.w;

        v16bf bcur = *(const v16bf*)(&Bt[(lane & 15) * 32 + kh16]);
#pragma unroll
        for (int j = 0; j < 8; ++j) {
            v16bf bnext = bcur;
            if (j < 7)
                bnext = *(const v16bf*)(&Bt[((j + 1) * 16 + (lane & 15)) * 32 + kh16]);
            acc[j] = wmma_bf16(a, bcur, acc[j]);
            bcur = bnext;
        }
    };

    stage(0, 0);
    wait_async0();
    __syncthreads();

#pragma unroll 1
    for (int k0 = 0; k0 < EMB; k0 += 64) {       // 12 iterations, parity static
        if (k0 + 32 < EMB) stage(k0 + 32, 1);
        compute(k0, &ldsB[0][0]);
        wait_async0();
        __syncthreads();

        if (k0 + 64 < EMB) stage(k0 + 64, 0);
        compute(k0 + 32, &ldsB[1][0]);
        wait_async0();
        __syncthreads();
    }

    // Epilogue: bias + decode c = h*288 + d*3 + {q,k,v}; scatter bf16.
#pragma unroll
    for (int j = 0; j < 8; ++j) {
        int   c     = n0 + j * 16 + (lane & 15);
        int   h     = c / 288;
        int   rem   = c - h * 288;
        int   d     = rem / 3;
        int   which = rem - d * 3;
        float bias  = bqkv[c];
#pragma unroll
        for (int r = 0; r < 8; ++r) {
            int m = m0 + r + 8 * (lane >> 4);
            int b_ = m >> 11;             // / SEQ
            int n  = m & (SEQ - 1);
            __bf16 bv = (__bf16)(acc[j][r] + bias);
            size_t bh = (size_t)b_ * NHEAD + h;
            if (which == 0)      Qb[(bh * SEQ + n) * HDIM + d] = bv;
            else if (which == 1) Kb[(bh * SEQ + n) * HDIM + d] = bv;
            else                 Vt[(bh * HDIM + d) * SEQ + n] = bv;
        }
    }
}

// ---------------------------------------------------------------------------
// Kernel 2: flash attention.  Block = 8 waves sharing one (b,h); each wave
// owns a 16-query tile; 32-key tiles of K and V double-buffered in LDS via
// async global->LDS loads.  Online softmax in f32 on the C/D fragment layout;
// inv_scale applied AFTER softmax (faithful to the reference).
// Output written bf16 to Aout[b, n, h*96 + d]  ("b h n d -> b n (h d)").
// ---------------------------------------------------------------------------
__global__ __launch_bounds__(256)
void flash_attn_kernel(const __bf16* __restrict__ Qb,
                       const __bf16* __restrict__ Kb,
                       const __bf16* __restrict__ Vt,
                       __bf16* __restrict__ Aout) {
    __shared__ __align__(32) __bf16 ldsK[2][32 * HDIM];   // [buf][key][d] 2x6 KB
    __shared__ __align__(32) __bf16 ldsV[2][HDIM * 32];   // [buf][d][key] 2x6 KB
    __shared__ __align__(32) __bf16 ldsP[8][16 * 32];     // per-wave P    8 KB

    const int lane = threadIdx.x & 31;
    const int wave = threadIdx.x >> 5;
    const int bh   = blockIdx.x >> 4;              // 0..31
    const int qblk = blockIdx.x & 15;
    const int b    = bh >> 3;
    const int h    = bh & 7;
    const int m0   = qblk * 128 + wave * 16;       // query tile base

    const __bf16* Qbase = Qb + ((size_t)bh * SEQ + m0) * HDIM;
    const __bf16* Kbase = Kb + (size_t)bh * SEQ * HDIM;
    const __bf16* Vbase = Vt + (size_t)bh * HDIM * SEQ;

    const int arow  = lane & 15;
    const int khalf = (lane >> 4) * 8;
    const int kh16  = (lane >> 4) * 16;

    // Straight-line staging: 384 16B chunks over 256 threads for K and V.
    const int tid = threadIdx.x;
    auto stage = [&](int kb, int p) {
        {   // K tile: 32 keys x 96 d, [key][d]
            int kr = tid / 12, kc = tid % 12;
            async_ld_b128(&ldsK[p][kr * HDIM + kc * 8],
                          Kbase + (size_t)(kb + kr) * HDIM + kc * 8);
            if (tid < 128) {
                int i2 = tid + 256;
                int kr2 = i2 / 12, kc2 = i2 % 12;
                async_ld_b128(&ldsK[p][kr2 * HDIM + kc2 * 8],
                              Kbase + (size_t)(kb + kr2) * HDIM + kc2 * 8);
            }
        }
        {   // V tile: 96 d x 32 keys, [d][key]
            int vr = tid >> 2, vc = tid & 3;
            async_ld_b128(&ldsV[p][vr * 32 + vc * 8],
                          Vbase + (size_t)vr * SEQ + kb + vc * 8);
            if (tid < 128) {
                int i2 = tid + 256;
                int vr2 = i2 >> 2, vc2 = i2 & 3;
                async_ld_b128(&ldsV[p][vr2 * 32 + vc2 * 8],
                              Vbase + (size_t)vr2 * SEQ + kb + vc2 * 8);
            }
        }
    };

    // Q A-fragments for the 3 d-slices of 32 (resident for whole kernel).
    v16bf qf[3];
    {
        const __bf16* qp = Qbase + (size_t)arow * HDIM;
#pragma unroll
        for (int c = 0; c < 3; ++c) {
            v8bf lo = *(const v8bf*)(qp + c * 32 + kh16);
            v8bf hi = *(const v8bf*)(qp + c * 32 + kh16 + 8);
#pragma unroll
            for (int e = 0; e < 8; ++e) { qf[c][e] = lo[e]; qf[c][8 + e] = hi[e]; }
        }
    }

    float mrow[8], lrow[8];
    v8f   acc[6] = {};
#pragma unroll
    for (int r = 0; r < 8; ++r) { mrow[r] = -1e30f; lrow[r] = 0.0f; }

    auto compute = [&](const __bf16* Kt, const __bf16* Vtile) {
        // S = Q @ K^T : two 16-key tiles, 3 d-slices each.
        v8f s[2] = {};
#pragma unroll
        for (int t = 0; t < 2; ++t) {
#pragma unroll
            for (int c = 0; c < 3; ++c) {
                v16bf bk = *(const v16bf*)(
                    &Kt[(t * 16 + (lane & 15)) * HDIM + c * 32 + kh16]);
                s[t] = wmma_bf16(qf[c], bk, s[t]);
            }
        }

        // Online softmax (row = r + 8*(lane>>4); cols across 16-lane half).
        float scale[8];
#pragma unroll
        for (int r = 0; r < 8; ++r) {
            float mx = fmaxf(s[0][r], s[1][r]);
            mx = fmaxf(mx, __shfl_xor(mx, 1));
            mx = fmaxf(mx, __shfl_xor(mx, 2));
            mx = fmaxf(mx, __shfl_xor(mx, 4));
            mx = fmaxf(mx, __shfl_xor(mx, 8));
            float mnew = fmaxf(mrow[r], mx);
            scale[r] = __expf(mrow[r] - mnew);
            mrow[r]  = mnew;
            float p0 = __expf(s[0][r] - mnew);
            float p1 = __expf(s[1][r] - mnew);
            s[0][r] = p0; s[1][r] = p1;
            float rs = p0 + p1;
            rs += __shfl_xor(rs, 1);
            rs += __shfl_xor(rs, 2);
            rs += __shfl_xor(rs, 4);
            rs += __shfl_xor(rs, 8);
            lrow[r] = lrow[r] * scale[r] + rs;
        }
#pragma unroll
        for (int c = 0; c < 6; ++c)
#pragma unroll
            for (int r = 0; r < 8; ++r)
                acc[c][r] = acc[c][r] * scale[r];

        // D-fragment -> A-fragment conversion of P via per-wave LDS tile.
        __bf16* P = &ldsP[wave][0];
#pragma unroll
        for (int t = 0; t < 2; ++t)
#pragma unroll
            for (int r = 0; r < 8; ++r)
                P[(r + 8 * (lane >> 4)) * 32 + t * 16 + (lane & 15)] =
                    (__bf16)s[t][r];
        __builtin_amdgcn_wave_barrier();   // in-order DS within wave

        v16bf pa;
        {
            v8bf lo = *(const v8bf*)(&P[arow * 32 + kh16]);
            v8bf hi = *(const v8bf*)(&P[arow * 32 + kh16 + 8]);
#pragma unroll
            for (int e = 0; e < 8; ++e) { pa[e] = lo[e]; pa[8 + e] = hi[e]; }
        }

        // O += P @ V : 6 d-chunks of 16.
#pragma unroll
        for (int c = 0; c < 6; ++c) {
            v16bf vb = *(const v16bf*)(&Vtile[(c * 16 + (lane & 15)) * 32 + kh16]);
            acc[c] = wmma_bf16(pa, vb, acc[c]);
        }
    };

    stage(0, 0);
    wait_async0();
    __syncthreads();

#pragma unroll 1
    for (int kb = 0; kb < SEQ; kb += 64) {       // 32 iterations, parity static
        if (kb + 32 < SEQ) stage(kb + 32, 1);
        compute(&ldsK[0][0], &ldsV[0][0]);
        wait_async0();
        __syncthreads();

        if (kb + 64 < SEQ) stage(kb + 64, 0);
        compute(&ldsK[1][0], &ldsV[1][0]);
        wait_async0();
        __syncthreads();
    }

    // Epilogue: (1/sqrt(96)) applied AFTER softmax, per the reference.
    const float inv_scale = 0.10206207261596577f;
#pragma unroll
    for (int r = 0; r < 8; ++r) {
        float sc = inv_scale / lrow[r];
        int   n  = m0 + r + 8 * (lane >> 4);
        __bf16* orow = Aout + ((size_t)b * SEQ + n) * EMB + h * HDIM;
#pragma unroll
        for (int c = 0; c < 6; ++c)
            orow[c * 16 + (lane & 15)] = (__bf16)(acc[c][r] * sc);
    }
}

// ---------------------------------------------------------------------------
// Kernel 3: output projection.  Out[8192,768] = A_bf16 @ Wproj + b_proj (f32)
// ---------------------------------------------------------------------------
__global__ __launch_bounds__(256)
void proj_gemm_kernel(const __bf16* __restrict__ A,    // [8192][768] bf16
                      const __bf16* __restrict__ Wt,   // [768][768]  bf16
                      const float* __restrict__ bias,
                      float* __restrict__ Out) {
    __shared__ __align__(32) __bf16 ldsB[2][128 * 32];

    const int lane  = threadIdx.x & 31;
    const int wave  = threadIdx.x >> 5;
    const int n0    = blockIdx.x * 128;
    const int m0    = (blockIdx.y * 8 + wave) * 16;
    const int arow  = m0 + (lane & 15);
    const int khalf = (lane >> 4) * 8;
    const int kh16  = (lane >> 4) * 16;

    const int sn = threadIdx.x >> 1;
    const int sk = (threadIdx.x & 1) * 16;

    auto stage = [&](int k0, int p) {
        const __bf16* src = Wt + (size_t)(n0 + sn) * EMB + k0 + sk;
        __bf16*       dst = &ldsB[p][sn * 32 + sk];
        async_ld_b128(dst, src);
        async_ld_b128(dst + 8, src + 8);
    };

    v8f acc[8] = {};

    auto compute = [&](int k0, const __bf16* Bt) {
        v16bf a;
        const __bf16* ap = A + (size_t)arow * EMB + k0;
        v8bf lo = *(const v8bf*)(ap + kh16);
        v8bf hi = *(const v8bf*)(ap + kh16 + 8);
#pragma unroll
        for (int e = 0; e < 8; ++e) { a[e] = lo[e]; a[8 + e] = hi[e]; }

        v16bf bcur = *(const v16bf*)(&Bt[(lane & 15) * 32 + kh16]);
#pragma unroll
        for (int j = 0; j < 8; ++j) {
            v16bf bnext = bcur;
            if (j < 7)
                bnext = *(const v16bf*)(&Bt[((j + 1) * 16 + (lane & 15)) * 32 + kh16]);
            acc[j] = wmma_bf16(a, bcur, acc[j]);
            bcur = bnext;
        }
    };

    stage(0, 0);
    wait_async0();
    __syncthreads();

#pragma unroll 1
    for (int k0 = 0; k0 < EMB; k0 += 64) {
        if (k0 + 32 < EMB) stage(k0 + 32, 1);
        compute(k0, &ldsB[0][0]);
        wait_async0();
        __syncthreads();

        if (k0 + 64 < EMB) stage(k0 + 64, 0);
        compute(k0 + 32, &ldsB[1][0]);
        wait_async0();
        __syncthreads();
    }

#pragma unroll
    for (int j = 0; j < 8; ++j) {
        int   c  = n0 + j * 16 + (lane & 15);
        float bb = bias[c];
#pragma unroll
        for (int r = 0; r < 8; ++r) {
            int m = m0 + r + 8 * (lane >> 4);
            Out[(size_t)m * EMB + c] = acc[j][r] + bb;
        }
    }
}

// ---------------------------------------------------------------------------
// Launch
// ---------------------------------------------------------------------------
extern "C" void kernel_launch(void* const* d_in, const int* in_sizes, int n_in,
                              void* d_out, int out_size, void* d_ws, size_t ws_size,
                              hipStream_t stream) {
    const float* x      = (const float*)d_in[0];
    const float* w_qkv  = (const float*)d_in[1];
    const float* b_qkv  = (const float*)d_in[2];
    const float* w_proj = (const float*)d_in[3];
    const float* b_proj = (const float*)d_in[4];
    float*       out    = (float*)d_out;

    // Workspace carve-up (~53 MB total).
    char*  ws  = (char*)d_ws;
    size_t off = 0;
    auto carve = [&](size_t bytes) -> char* {
        char* p = ws + off;
        off += (bytes + 255) & ~size_t(255);
        return p;
    };
    const size_t qkv_bytes = (size_t)BATCH * NHEAD * SEQ * HDIM * sizeof(__bf16); // 12.58 MB
    __bf16* Qb     = (__bf16*)carve(qkv_bytes);
    __bf16* Kb     = (__bf16*)carve(qkv_bytes);
    __bf16* Vt     = (__bf16*)carve(qkv_bytes);
    __bf16* Aattn  = (__bf16*)carve((size_t)MROWS * EMB * sizeof(__bf16));        // 12.58 MB
    __bf16* Wqkvt  = (__bf16*)carve((size_t)QKV_N * EMB * sizeof(__bf16));        //  3.54 MB
    __bf16* Wprojt = (__bf16*)carve((size_t)EMB * EMB * sizeof(__bf16));          //  1.18 MB

    repack_w_kernel<<<(QKV_N * EMB + 255) / 256, 256, 0, stream>>>(w_qkv, Wqkvt, EMB, QKV_N);
    repack_w_kernel<<<(EMB * EMB + 255) / 256, 256, 0, stream>>>(w_proj, Wprojt, EMB, EMB);

    qkv_gemm_kernel<<<dim3(QKV_N / 128, MROWS / 128), 256, 0, stream>>>(
        x, b_qkv, Wqkvt, Qb, Kb, Vt);

    flash_attn_kernel<<<BATCH * NHEAD * (SEQ / 128), 256, 0, stream>>>(
        Qb, Kb, Vt, Aattn);

    proj_gemm_kernel<<<dim3(EMB / 128, MROWS / 128), 256, 0, stream>>>(
        Aattn, Wprojt, b_proj, out);
}
